// MoELayer_26268019982646
// MI455X (gfx1250) — matrix operand order repaired and checked by reading, not compiled, not run on hardware
//
#include <hip/hip_runtime.h>

// Problem constants (BERT-base MoE, gate-token routing)
#define N_TOK 4096          // B*S = 8*512
#define DIM   768
#define FDIM  3072
#define NEXP  4
#define LDA   34            // LDS row stride (ushorts) for A tiles: 17 banks/row, coprime 64
#define LDB   130           // LDS row stride (ushorts) for B tiles: 65 banks/row, coprime 64

// Probe the CDNA5 async global->LDS copy path (guarded; falls back to reg staging)
#if defined(__HIP_DEVICE_COMPILE__) && __has_builtin(__builtin_amdgcn_global_load_async_to_lds_b128)
#define USE_ASYNC 1
#else
#define USE_ASYNC 0
#endif

typedef __attribute__((ext_vector_type(16))) __bf16 v16bf;
typedef __attribute__((ext_vector_type(8)))  float  v8f;
typedef __attribute__((ext_vector_type(4)))  int    v4i_;

union FragU { unsigned short us[16]; v16bf v; };

// fp32 -> bf16 round-to-nearest-even (bit math, no header types)
__device__ __forceinline__ unsigned short f2bf(float f) {
  unsigned int x = __float_as_uint(f);
  x += 0x7fffu + ((x >> 16) & 1u);
  return (unsigned short)(x >> 16);
}

__device__ __forceinline__ v8f wmma_bf16(v16bf a, v16bf b, v8f c) {
  // (neg_a, A, neg_b, B, c_mod, C, reuse_a, reuse_b)
  return __builtin_amdgcn_wmma_f32_16x16x32_bf16(false, a, false, b, (short)0, c,
                                                 false, false);
}

// A-matrix 16x32 bf16 fragment per ISA layout:
// lanes 0-15 row M=lane, K in {0..7,16..23}; lanes 16-31 same row, K in {8..15,24..31}
__device__ __forceinline__ v16bf load_a_frag(const unsigned short* As, int mr0, int lane) {
  FragU f;
  const unsigned short* row = As + (mr0 + (lane & 15)) * LDA;
  const int hb = (lane >> 4) * 8;
  #pragma unroll
  for (int j = 0; j < 8; ++j) {
    const int kk = ((j < 4) ? (2 * j) : (8 + 2 * j)) + hb;  // 0,2,4,6 / 16,18,20,22 (+8)
    f.us[2 * j]     = row[kk];
    f.us[2 * j + 1] = row[kk + 1];
  }
  return f.v;
}

// B-matrix 32x16 bf16 fragment: lane%16 = N column; lanes 0-15 K in {0..7,16..23}, lanes 16-31 K in {8..15,24..31}
__device__ __forceinline__ v16bf load_b_frag(const unsigned short* Bs, int nc0, int lane) {
  FragU f;
  const unsigned short* col = Bs + nc0 + (lane & 15);
  const int hb = (lane >> 4) * 8;
  #pragma unroll
  for (int j = 0; j < 16; ++j) f.us[j] = col[((j < 8) ? j : (8 + j)) + hb) * LDB];
  return f.v;
}

__device__ __forceinline__ void async_wait0() {
#if USE_ASYNC
#if __has_builtin(__builtin_amdgcn_s_wait_asynccnt)
  __builtin_amdgcn_s_wait_asynccnt(0);
#else
  asm volatile("s_wait_asynccnt 0x0" ::: "memory");
#endif
#endif
}

#if USE_ASYNC
// builtin wants (global int4*, lds int4*, imm offset, imm cpol) per hipcc diagnostic
typedef __attribute__((address_space(1))) v4i_ gv4i;
typedef __attribute__((address_space(3))) v4i_ lv4i;
__device__ __forceinline__ void async_cp_b128(const void* gsrc, void* ldst) {
  __builtin_amdgcn_global_load_async_to_lds_b128((gv4i*)gsrc, (lv4i*)ldst, 0, 0);
}
#endif

// ---------------------------------------------------------------------------
// Kernel 1: gating. One wave32 per token.
// ---------------------------------------------------------------------------
__global__ __launch_bounds__(256) void gate_kernel(
    const float* __restrict__ x, const float* __restrict__ gw,
    float* __restrict__ prob, int* __restrict__ eid,
    int* __restrict__ posArr, int* __restrict__ counts) {
  const int wave = threadIdx.x >> 5;
  const int lane = threadIdx.x & 31;
  const int n = blockIdx.x * 8 + wave;
  const float* xr = x + (size_t)n * DIM;
  float a0 = 0.f, a1 = 0.f, a2 = 0.f, a3 = 0.f;
  for (int d = lane; d < DIM; d += 32) {
    const float xv = xr[d];
    a0 += xv * gw[0 * DIM + d];
    a1 += xv * gw[1 * DIM + d];
    a2 += xv * gw[2 * DIM + d];
    a3 += xv * gw[3 * DIM + d];
  }
  #pragma unroll
  for (int off = 16; off > 0; off >>= 1) {
    a0 += __shfl_xor(a0, off, 32);
    a1 += __shfl_xor(a1, off, 32);
    a2 += __shfl_xor(a2, off, 32);
    a3 += __shfl_xor(a3, off, 32);
  }
  if (lane == 0) {
    float lg[4] = {a0, a1, a2, a3};
    float mx = lg[0]; int am = 0;
    #pragma unroll
    for (int e = 1; e < NEXP; ++e) if (lg[e] > mx) { mx = lg[e]; am = e; }  // first-index ties
    float pr[4], s = 0.f;
    #pragma unroll
    for (int e = 0; e < NEXP; ++e) { pr[e] = __expf(lg[e] - mx); s += pr[e]; }
    const float inv = 1.f / s;
    #pragma unroll
    for (int e = 0; e < NEXP; ++e) prob[(size_t)n * NEXP + e] = pr[e] * inv;
    eid[n]    = am;
    posArr[n] = atomicAdd(&counts[am], 1);
  }
}

// ---------------------------------------------------------------------------
// Kernel 2: deterministic balance-loss reduction + offsets + scalar outputs.
// ---------------------------------------------------------------------------
__global__ __launch_bounds__(256) void finalize_kernel(
    const float* __restrict__ prob, int* __restrict__ ctrl,  // [0..3]=counts [4..7]=offs
    float* __restrict__ out_tail) {                          // d_out + N_TOK*DIM
  __shared__ float red[NEXP * 256];
  const int tid = threadIdx.x;
  float s[NEXP] = {0.f, 0.f, 0.f, 0.f};
  for (int n = tid; n < N_TOK; n += 256) {
    #pragma unroll
    for (int e = 0; e < NEXP; ++e) s[e] += prob[(size_t)n * NEXP + e];
  }
  #pragma unroll
  for (int e = 0; e < NEXP; ++e) red[e * 256 + tid] = s[e];
  __syncthreads();
  for (int step = 128; step > 0; step >>= 1) {
    if (tid < step) {
      #pragma unroll
      for (int e = 0; e < NEXP; ++e) red[e * 256 + tid] += red[e * 256 + tid + step];
    }
    __syncthreads();
  }
  if (tid == 0) {
    const float invN = 1.f / (float)N_TOK;
    float loss = 0.f;
    int off = 0;
    #pragma unroll
    for (int e = 0; e < NEXP; ++e) {
      const int c = ctrl[e];
      loss += (red[e * 256] * invN) * ((float)c * invN);
      ctrl[4 + e] = off;              // exclusive scan -> slot offsets
      off += c;
      out_tail[1 + e] = (float)c;     // gate_load (as float values)
    }
    out_tail[0] = (float)NEXP * loss; // balance_loss
  }
}

// ---------------------------------------------------------------------------
// Kernel 3: build token_of_slot permutation (grouped-GEMM gather list).
// ---------------------------------------------------------------------------
__global__ __launch_bounds__(256) void scatter_kernel(
    const int* __restrict__ eid, const int* __restrict__ posArr,
    const int* __restrict__ ctrl, int* __restrict__ tos) {
  const int n = blockIdx.x * 256 + threadIdx.x;
  tos[ctrl[4 + eid[n]] + posArr[n]] = n;
}

// ---------------------------------------------------------------------------
// Kernel 4: grouped GEMM1 + exact GELU.  H[slot, F] = gelu(Xg @ W1[e] + b1[e])
// 128x128 tiles, K-steps of 32, double-buffered LDS, bf16 WMMA / f32 acc.
// ---------------------------------------------------------------------------
__global__ __launch_bounds__(256) void ffn1_kernel(
    const float* __restrict__ x, const float* __restrict__ W1,
    const float* __restrict__ b1, const int* __restrict__ tos,
    const int* __restrict__ ctrl, unsigned short* __restrict__ hbuf) {
  const int e   = blockIdx.z;
  const int cnt = ctrl[e];
  const int m0  = blockIdx.y * 128;
  if (m0 >= cnt) return;
  const int rowsHere = (cnt - m0 < 128) ? (cnt - m0) : 128;
  const int slotBase = ctrl[4 + e] + m0;
  const int n0 = blockIdx.x * 128;

  __shared__ unsigned short As[2][128 * LDA];
  __shared__ unsigned short Bs[2][32 * LDB];

  const int tid = threadIdx.x, lane = tid & 31, wave = tid >> 5;
  const int ar = tid >> 1, ah = tid & 1;            // A staging: row / 16-col half
  const int bk = tid >> 3, bc = (tid & 7) * 16;     // B staging: K row / col base
  const int tokenA = (ar < rowsHere) ? tos[slotBase + ar] : -1;
  const float* wbase = W1 + (size_t)e * DIM * FDIM;

  // zero-fill inactive A rows ONCE (both buffers); never overwritten later
  if (tokenA < 0) {
    #pragma unroll
    for (int b = 0; b < 2; ++b) {
      unsigned int* dst = (unsigned int*)&As[b][ar * LDA + ah * 16];
      #pragma unroll
      for (int i = 0; i < 8; ++i) dst[i] = 0u;
    }
  }

  float4 ra[4], rb[4];
  auto loadA = [&](int k0) {
    if (tokenA >= 0) {
      const float4* s = (const float4*)(x + (size_t)tokenA * DIM + k0 + ah * 16);
      ra[0] = s[0]; ra[1] = s[1]; ra[2] = s[2]; ra[3] = s[3];
    }
  };
  auto loadB = [&](int k0) {
    const float4* s = (const float4*)(wbase + (size_t)(k0 + bk) * FDIM + n0 + bc);
    rb[0] = s[0]; rb[1] = s[1]; rb[2] = s[2]; rb[3] = s[3];
  };
  auto storeA = [&](int b) {
    if (tokenA >= 0) {
      unsigned int* dst = (unsigned int*)&As[b][ar * LDA + ah * 16];
      #pragma unroll
      for (int i = 0; i < 4; ++i) {
        dst[2 * i]     = ((unsigned)f2bf(ra[i].y) << 16) | f2bf(ra[i].x);
        dst[2 * i + 1] = ((unsigned)f2bf(ra[i].w) << 16) | f2bf(ra[i].z);
      }
    }
  };
  auto storeB = [&](int b) {
    unsigned int* dst = (unsigned int*)&Bs[b][bk * LDB + bc];
    #pragma unroll
    for (int i = 0; i < 4; ++i) {
      dst[2 * i]     = ((unsigned)f2bf(rb[i].y) << 16) | f2bf(rb[i].x);
      dst[2 * i + 1] = ((unsigned)f2bf(rb[i].w) << 16) | f2bf(rb[i].z);
    }
  };

  const int tr0 = (wave >> 2) * 4;                  // 4 tile rows per wave
  const int tc0 = (wave & 3) * 2;                   // 2 tile cols per wave
  v8f C[4][2];
  #pragma unroll
  for (int i = 0; i < 4; ++i)
    #pragma unroll
    for (int c = 0; c < 2; ++c) C[i][c] = (v8f){0.f, 0.f, 0.f, 0.f, 0.f, 0.f, 0.f, 0.f};

  loadA(0); loadB(0); storeA(0); storeB(0);
  int buf = 0;
  for (int k0 = 0; k0 < DIM; k0 += 32) {
    __syncthreads();
    const int kn = k0 + 32;
    if (kn < DIM) { loadA(kn); loadB(kn); }         // overlap next-tile loads with WMMA
    const unsigned short* Ab = As[buf];
    const unsigned short* Bb = Bs[buf];
    v16bf bfr0 = load_b_frag(Bb, (tc0 + 0) * 16, lane);
    v16bf bfr1 = load_b_frag(Bb, (tc0 + 1) * 16, lane);
    #pragma unroll
    for (int i = 0; i < 4; ++i) {
      v16bf afr = load_a_frag(Ab, (tr0 + i) * 16, lane);
      C[i][0] = wmma_bf16(afr, bfr0, C[i][0]);
      C[i][1] = wmma_bf16(afr, bfr1, C[i][1]);
    }
    if (kn < DIM) { storeA(buf ^ 1); storeB(buf ^ 1); }
    buf ^= 1;
  }

  // epilogue: +bias, exact GELU, store bf16 H
  const int r16 = lane & 15, hi8 = (lane >> 4) * 8;
  #pragma unroll
  for (int i = 0; i < 4; ++i) {
    #pragma unroll
    for (int c = 0; c < 2; ++c) {
      const int col  = n0 + (tc0 + c) * 16 + r16;
      const float bv = b1[e * FDIM + col];
      #pragma unroll
      for (int j = 0; j < 8; ++j) {
        const int lr = (tr0 + i) * 16 + j + hi8;
        if (lr < rowsHere) {
          float v = C[i][c][j] + bv;
          v = 0.5f * v * (1.0f + erff(v * 0.70710678118654752f));   // approximate=False
          hbuf[(size_t)(slotBase + lr) * FDIM + col] = f2bf(v);
        }
      }
    }
  }
}

// ---------------------------------------------------------------------------
// Kernel 5: grouped GEMM2 + scatter.  out[token] = H @ W2[e] + b2[e]
// A tile is already bf16 -> use CDNA5 async global->LDS copy when available.
// ---------------------------------------------------------------------------
__global__ __launch_bounds__(256) void ffn2_kernel(
    const unsigned short* __restrict__ hbuf, const float* __restrict__ W2,
    const float* __restrict__ b2, const int* __restrict__ tos,
    const int* __restrict__ ctrl, float* __restrict__ out) {
  const int e   = blockIdx.z;
  const int cnt = ctrl[e];
  const int m0  = blockIdx.y * 128;
  if (m0 >= cnt) return;
  const int rowsHere = (cnt - m0 < 128) ? (cnt - m0) : 128;
  const int slotBase = ctrl[4 + e] + m0;
  const int n0 = blockIdx.x * 128;

  __shared__ unsigned short As[2][128 * LDA];
  __shared__ unsigned short Bs[2][32 * LDB];

  const int tid = threadIdx.x, lane = tid & 31, wave = tid >> 5;
  const int ar = tid >> 1, ah = tid & 1;
  const int bk = tid >> 3, bc = (tid & 7) * 16;
  const bool activeA = (ar < rowsHere);
  const unsigned short* asrc =
      hbuf + (size_t)(slotBase + (activeA ? ar : 0)) * FDIM + ah * 16;
  const float* wbase = W2 + (size_t)e * FDIM * DIM;

  // zero-fill inactive A rows ONCE (both buffers)
  if (!activeA) {
    #pragma unroll
    for (int b = 0; b < 2; ++b) {
      unsigned int* dst = (unsigned int*)&As[b][ar * LDA + ah * 16];
      #pragma unroll
      for (int i = 0; i < 8; ++i) dst[i] = 0u;
    }
  }

  float4 rb[4];
#if USE_ASYNC
  auto asyncA = [&](int k0, int b) {                // raw bf16 copy: async DMA to LDS
    if (activeA) {
      unsigned short* d = &As[b][ar * LDA + ah * 16];
      async_cp_b128(asrc + k0,     d);
      async_cp_b128(asrc + k0 + 8, d + 8);
    }
  };
#else
  uint4 raw0, raw1;
  auto loadA = [&](int k0) {
    if (activeA) {
      raw0 = *(const uint4*)(asrc + k0);
      raw1 = *(const uint4*)(asrc + k0 + 8);
    }
  };
  auto storeA = [&](int b) {
    if (activeA) {
      unsigned int* dst = (unsigned int*)&As[b][ar * LDA + ah * 16];
      dst[0] = raw0.x; dst[1] = raw0.y; dst[2] = raw0.z; dst[3] = raw0.w;
      dst[4] = raw1.x; dst[5] = raw1.y; dst[6] = raw1.z; dst[7] = raw1.w;
    }
  };
#endif
  auto loadB = [&](int k0) {
    const float4* s = (const float4*)(wbase + (size_t)(k0 + bk) * DIM + n0 + bc);
    rb[0] = s[0]; rb[1] = s[1]; rb[2] = s[2]; rb[3] = s[3];
  };
  auto storeB = [&](int b) {
    unsigned int* dst = (unsigned int*)&Bs[b][bk * LDB + bc];
    #pragma unroll
    for (int i = 0; i < 4; ++i) {
      dst[2 * i]     = ((unsigned)f2bf(rb[i].y) << 16) | f2bf(rb[i].x);
      dst[2 * i + 1] = ((unsigned)f2bf(rb[i].w) << 16) | f2bf(rb[i].z);
    }
  };

  const int tr0 = (wave >> 2) * 4;
  const int tc0 = (wave & 3) * 2;
  v8f C[4][2];
  #pragma unroll
  for (int i = 0; i < 4; ++i)
    #pragma unroll
    for (int c = 0; c < 2; ++c) C[i][c] = (v8f){0.f, 0.f, 0.f, 0.f, 0.f, 0.f, 0.f, 0.f};

#if USE_ASYNC
  asyncA(0, 0);
#else
  loadA(0); storeA(0);
#endif
  loadB(0); storeB(0);
  int buf = 0;
  for (int k0 = 0; k0 < FDIM; k0 += 32) {
    async_wait0();                                  // this wave's async writes done
    __syncthreads();                                // all waves' LDS writes visible
    const int kn = k0 + 32;
    if (kn < FDIM) {
#if USE_ASYNC
      asyncA(kn, buf ^ 1);
#else
      loadA(kn);
#endif
      loadB(kn);
    }
    const unsigned short* Ab = As[buf];
    const unsigned short* Bb = Bs[buf];
    v16bf bfr0 = load_b_frag(Bb, (tc0 + 0) * 16, lane);
    v16bf bfr1 = load_b_frag(Bb, (tc0 + 1) * 16, lane);
    #pragma unroll
    for (int i = 0; i < 4; ++i) {
      v16bf afr = load_a_frag(Ab, (tr0 + i) * 16, lane);
      C[i][0] = wmma_bf16(afr, bfr0, C[i][0]);
      C[i][1] = wmma_bf16(afr, bfr1, C[i][1]);
    }
    if (kn < FDIM) {
#if !USE_ASYNC
      storeA(buf ^ 1);
#endif
      storeB(buf ^ 1);
    }
    buf ^= 1;
  }

  // epilogue: +bias, scatter rows back to original token positions (fp32)
  const int r16 = lane & 15, hi8 = (lane >> 4) * 8;
  #pragma unroll
  for (int i = 0; i < 4; ++i) {
    #pragma unroll
    for (int c = 0; c < 2; ++c) {
      const int col  = n0 + (tc0 + c) * 16 + r16;
      const float bv = b2[e * DIM + col];
      #pragma unroll
      for (int j = 0; j < 8; ++j) {
        const int lr = (tr0 + i) * 16 + j + hi8;
        if (lr < rowsHere) {
          const int token = tos[slotBase + lr];
          out[(size_t)token * DIM + col] = C[i][c][j] + bv;
        }
      }
    }
  }
}

// ---------------------------------------------------------------------------
extern "C" void kernel_launch(void* const* d_in, const int* in_sizes, int n_in,
                              void* d_out, int out_size, void* d_ws, size_t ws_size,
                              hipStream_t stream) {
  // setup_inputs order: x, input_ids, attention_mask, gate_w, W1, b1, W2, b2
  const float* x  = (const float*)d_in[0];
  const float* gw = (const float*)d_in[3];
  const float* W1 = (const float*)d_in[4];
  const float* b1 = (const float*)d_in[5];
  const float* W2 = (const float*)d_in[6];
  const float* b2 = (const float*)d_in[7];
  float* out = (float*)d_out;                 // [N*D] out, [ND] loss, [ND+1..4] gate_load

  // workspace carve (needs ~25.3 MB)
  char* w = (char*)d_ws;
  float* prob  = (float*)(w);                          // 4096*4*4   = 64 KB
  int*   eid   = (int*)(w + (64 << 10));               // 16 KB
  int*   posA  = (int*)(w + (80 << 10));               // 16 KB
  int*   tos   = (int*)(w + (96 << 10));               // 16 KB
  int*   ctrl  = (int*)(w + (112 << 10));              // counts[4] + offs[4]
  unsigned short* hbuf = (unsigned short*)(w + (128 << 10)); // N*F bf16 = 24 MB

  (void)hipMemsetAsync(ctrl, 0, 8 * sizeof(int), stream); // zero routing counters

  gate_kernel<<<N_TOK / 8, 256, 0, stream>>>(x, gw, prob, eid, posA, ctrl);
  finalize_kernel<<<1, 256, 0, stream>>>(prob, ctrl, out + (size_t)N_TOK * DIM);
  scatter_kernel<<<N_TOK / 256, 256, 0, stream>>>(eid, posA, ctrl, tos);

  dim3 g1(FDIM / 128, N_TOK / 128, NEXP);   // (24, 32, 4) — early-exit on counts
  ffn1_kernel<<<g1, 256, 0, stream>>>(x, W1, b1, tos, ctrl, hbuf);

  dim3 g2(DIM / 128, N_TOK / 128, NEXP);    // (6, 32, 4)
  ffn2_kernel<<<g2, 256, 0, stream>>>(hbuf, W2, b2, tos, ctrl, out);
}
